// GraphSAGE_Encoder_SubGraph_59425167507611
// MI455X (gfx1250) — matrix operand order repaired and checked by reading, not compiled, not run on hardware
//
#include <hip/hip_runtime.h>
#include <cstdint>

#define N_NODES 10000
#define D_INF   512
#define D_HID   256
#define NCLS    40
#define NCLS_P  48
#define N_EDGES 160000
#define NT_COL  625          // N_NODES / 16

typedef __attribute__((ext_vector_type(16))) _Float16 v16h;
typedef __attribute__((ext_vector_type(8)))  _Float16 v8h;
typedef __attribute__((ext_vector_type(8)))  float    v8f;

// A-fragment for v_wmma_f32_16x16x32_f16, wave32:
// lane l holds row M=l%16, K in [16s,16s+16), s=l/16.
__device__ __forceinline__ v16h load_a_frag(const _Float16* p) {
  v8h lo = *(const v8h*)p;
  v8h hi = *(const v8h*)(p + 8);
  return __builtin_shufflevector(lo, hi, 0,1,2,3,4,5,6,7,8,9,10,11,12,13,14,15);
}

// ---------------- graph structure ----------------
__global__ void zero_i32(int* p, int n) {
  int i = blockIdx.x * 256 + threadIdx.x;
  if (i < n) p[i] = 0;
}

__global__ void count_deg(const int* __restrict__ dst, int* __restrict__ cnt, int e) {
  int i = blockIdx.x * 256 + threadIdx.x;
  if (i < e) atomicAdd(&cnt[dst[i]], 1);
}

__global__ void scan10k(const int* __restrict__ cnt, int* __restrict__ rstart,
                        int* __restrict__ fpos, int n) {
  __shared__ int part[256];
  int t = threadIdx.x;
  int chunk = (n + 255) / 256;
  int b0 = t * chunk;
  int b1 = b0 + chunk; if (b1 > n) b1 = n;
  int s = 0;
  for (int i = b0; i < b1; ++i) s += cnt[i];
  part[t] = s;
  __syncthreads();
  if (t == 0) {
    int acc = 0;
    for (int i = 0; i < 256; ++i) { int v = part[i]; part[i] = acc; acc += v; }
    rstart[n] = acc;
  }
  __syncthreads();
  int acc = part[t];
  for (int i = b0; i < b1; ++i) { rstart[i] = acc; fpos[i] = acc; acc += cnt[i]; }
}

__global__ void fill_csr(const int* __restrict__ src, const int* __restrict__ dst,
                         int* __restrict__ fpos, int* __restrict__ csr, int e) {
  int i = blockIdx.x * 256 + threadIdx.x;
  if (i < e) {
    int p = atomicAdd(&fpos[dst[i]], 1);
    csr[p] = src[i];
  }
}

// ---------------- conversions ----------------
__global__ void f2h(const float* __restrict__ s, _Float16* __restrict__ d, int n) {
  int i = blockIdx.x * 256 + threadIdx.x;
  if (i < n) d[i] = (_Float16)s[i];
}

__global__ void pad_w(const float* __restrict__ W, _Float16* __restrict__ Wp,
                      int crows, int k, int total) {
  int i = blockIdx.x * 256 + threadIdx.x;
  if (i >= total) return;
  int r = i / k, kk = i % k;
  Wp[i] = (_Float16)(r < crows ? W[r * k + kk] : 0.0f);
}

__global__ void pad_b(const float* __restrict__ b, float* __restrict__ bp, int c, int cp) {
  int i = threadIdx.x;
  if (i < cp) bp[i] = (i < c) ? b[i] : 0.0f;
}

// ---------------- segment mean (node-parallel, deterministic order) ----------------
__global__ __launch_bounds__(256) void aggregate(const float* __restrict__ X,
    const int* __restrict__ rstart, const int* __restrict__ csr,
    _Float16* __restrict__ outA, int F)
{
  __shared__ int lbuf[512];
  int n = blockIdx.x, t = threadIdx.x;
  int s0 = rstart[n], s1 = rstart[n + 1];
  int deg = s1 - s0;
  bool useL = (deg <= 512);
  if (useL) {
    for (int i = t; i < deg; i += 256) lbuf[i] = csr[s0 + i];
    __syncthreads();
    if (t == 0) {  // insertion sort -> deterministic summation order
      for (int i = 1; i < deg; ++i) {
        int key = lbuf[i]; int j = i - 1;
        while (j >= 0 && lbuf[j] > key) { lbuf[j + 1] = lbuf[j]; --j; }
        lbuf[j + 1] = key;
      }
    }
    __syncthreads();
  }
  float acc0 = 0.f, acc1 = 0.f;
  for (int e = 0; e < deg; ++e) {
    int sn = useL ? lbuf[e] : csr[s0 + e];
    const float* xr = X + (size_t)sn * F;
    acc0 += xr[t];
    if (F > 256) acc1 += xr[t + 256];
  }
  float inv = 1.0f / fmaxf((float)deg, 1.0f);
  outA[(size_t)n * F + t] = (_Float16)(acc0 * inv);
  if (F > 256) outA[(size_t)n * F + t + 256] = (_Float16)(acc1 * inv);
}

// ---------------- fused SAGE layer: ELU(A1@W1^T + A2@W2^T + b) via WMMA ----------------
__global__ __launch_bounds__(256) void conv_wmma(
    const _Float16* __restrict__ A1, const _Float16* __restrict__ W1,
    const _Float16* __restrict__ A2, const _Float16* __restrict__ W2,
    const float* __restrict__ bias, float* __restrict__ out,
    int Kdim, int ntn, int out_cols, int out_stride, int total_tiles)
{
  int wave = threadIdx.x >> 5;
  int lane = threadIdx.x & 31;
  int tile = blockIdx.x * 8 + wave;
  if (tile >= total_tiles) return;          // wave-uniform
  int mt = tile / ntn, nt = tile % ntn;
  int mbase = mt * 16, nbase = nt * 16;
  int r16 = lane & 15, s = lane >> 4;
  int arow = mbase + r16;
  int bcol = nbase + r16;
  float bv = bias[bcol];
  v8f c;
  #pragma unroll
  for (int i = 0; i < 8; ++i) c[i] = bv;
  const _Float16* a1p = A1 + (size_t)arow * Kdim + 16 * s;
  const _Float16* a2p = A2 + (size_t)arow * Kdim + 16 * s;
  const _Float16* w1p = W1 + (size_t)bcol * Kdim + 16 * s;
  const _Float16* w2p = W2 + (size_t)bcol * Kdim + 16 * s;
  for (int kb = 0; kb < Kdim; kb += 32) {
    v16h af1 = load_a_frag(a1p + kb);
    v16h bf1 = *(const v16h*)(w1p + kb);
    c = __builtin_amdgcn_wmma_f32_16x16x32_f16(false, af1, false, bf1, (short)0, c, false, false);
    v16h af2 = load_a_frag(a2p + kb);
    v16h bf2 = *(const v16h*)(w2p + kb);
    c = __builtin_amdgcn_wmma_f32_16x16x32_f16(false, af2, false, bf2, (short)0, c, false, false);
  }
  #pragma unroll
  for (int r = 0; r < 8; ++r) {
    int row = mbase + r + 8 * s;
    float v = c[r];
    v = (v > 0.f) ? v : (expf(v) - 1.0f);   // ELU
    if (bcol < out_cols) out[(size_t)row * out_stride + bcol] = v;
  }
}

// ---------------- row L2 norm -> f16 copies ----------------
__global__ __launch_bounds__(256) void rownorm(const float* __restrict__ H,
    _Float16* __restrict__ hh, _Float16* __restrict__ en)
{
  __shared__ float red[256];
  int row = blockIdx.x, t = threadIdx.x;
  float v = H[(size_t)row * D_HID + t];
  red[t] = v * v;
  __syncthreads();
  for (int st = 128; st > 0; st >>= 1) {
    if (t < st) red[t] += red[t + st];
    __syncthreads();
  }
  float nrm = fmaxf(sqrtf(red[0]), 1e-8f);
  hh[(size_t)row * D_HID + t] = (_Float16)v;
  en[(size_t)row * D_HID + t] = (_Float16)(v / nrm);
}

// ---------------- fused: en@en.T * mask -> streaming top-16 -> label fusion
//                  -> log_softmax blend with classifier ----------------
__global__ __launch_bounds__(256) void sim_topk(const _Float16* __restrict__ en,
    const float* __restrict__ mask, const int* __restrict__ y,
    const float* __restrict__ lc, float* __restrict__ out)
{
  __shared__ float tileS[8][16][17];
  __shared__ float mV[8][16][16];
  __shared__ int   mI[8][16][16];
  __shared__ float fuse[16][NCLS + 1];

  int wave = threadIdx.x >> 5, lane = threadIdx.x & 31;
  int r16 = lane & 15, s = lane >> 4;
  int rbase = blockIdx.x * 16;

  // keep the 16x256 A stripe resident in registers (8 k-steps)
  v16h afr[8];
  const _Float16* ap = en + (size_t)(rbase + r16) * D_HID + 16 * s;
  #pragma unroll
  for (int ks = 0; ks < 8; ++ks) afr[ks] = load_a_frag(ap + ks * 32);

  float best[16]; int bidx[16];
  #pragma unroll
  for (int q = 0; q < 16; ++q) { best[q] = -3.0e38f; bidx[q] = 0; }

  for (int jt = wave; jt < NT_COL; jt += 8) {
    int cbase = jt * 16;
    // prefetch next tile's mask rows (read-once stream, NT path below)
    int cnext = cbase + 8 * 16;
    if (cnext < N_NODES) {
      #pragma unroll
      for (int r = 0; r < 8; ++r) {
        int row = rbase + r + 8 * s;
        __builtin_prefetch(&mask[(size_t)row * N_NODES + cnext], 0, 0);
      }
    }
    const _Float16* bp = en + (size_t)(cbase + r16) * D_HID + 16 * s;
    v8f c;
    #pragma unroll
    for (int i = 0; i < 8; ++i) c[i] = 0.0f;
    #pragma unroll
    for (int ks = 0; ks < 8; ++ks) {
      v16h bf = *(const v16h*)(bp + ks * 32);
      c = __builtin_amdgcn_wmma_f32_16x16x32_f16(false, afr[ks], false, bf, (short)0, c, false, false);
    }
    // apply mask (non-temporal: 400 MB read-once; keep en resident in L2)
    // C layout: vgpr r -> row r+8s, col r16
    #pragma unroll
    for (int r = 0; r < 8; ++r) {
      int row = rbase + r + 8 * s;
      float mv = __builtin_nontemporal_load(&mask[(size_t)row * N_NODES + cbase + r16]);
      tileS[wave][r + 8 * s][r16] = c[r] * mv;
    }
    asm volatile("s_wait_dscnt 0" ::: "memory");   // cross-lane LDS RAW (wave-internal)
    __builtin_amdgcn_wave_barrier();
    if (s == 0) {            // lanes 0..15 each scan their row of the 16x16 tile
      #pragma unroll
      for (int c16 = 0; c16 < 16; ++c16) {
        float v = tileS[wave][lane][c16];
        if (v > best[15]) {
          best[15] = v; bidx[15] = cbase + c16;
          #pragma unroll
          for (int q = 15; q > 0; --q) {
            if (best[q] > best[q - 1]) {
              float tv = best[q]; best[q] = best[q - 1]; best[q - 1] = tv;
              int ti = bidx[q];  bidx[q] = bidx[q - 1]; bidx[q - 1] = ti;
            }
          }
        }
      }
    }
    __builtin_amdgcn_wave_barrier();
  }

  if (s == 0) {
    #pragma unroll
    for (int q = 0; q < 16; ++q) { mV[wave][lane][q] = best[q]; mI[wave][lane][q] = bidx[q]; }
  }
  __syncthreads();
  for (int i = threadIdx.x; i < 16 * (NCLS + 1); i += 256) (&fuse[0][0])[i] = 0.0f;
  __syncthreads();

  if (threadIdx.x < 16) {
    int row = threadIdx.x;
    int ptr[8];
    #pragma unroll
    for (int w = 0; w < 8; ++w) ptr[w] = 0;
    for (int t = 0; t < 16; ++t) {         // 8-way merge of sorted lists
      float bv = -3.4e38f; int bw = 0, bi = 0;
      #pragma unroll
      for (int w = 0; w < 8; ++w) {
        if (ptr[w] < 16) {
          float v = mV[w][row][ptr[w]];
          if (v > bv) { bv = v; bw = w; bi = mI[w][row][ptr[w]]; }
        }
      }
      #pragma unroll
      for (int w = 0; w < 8; ++w) if (w == bw) ptr[w]++;
      int cls = y[bi];
      fuse[row][cls] += expf(bv);          // exp-weighted one-hot fusion
    }
    // log_softmax(fused) and log_softmax(lc), blend with ETA=0.5
    float mx = -3.4e38f;
    for (int cc = 0; cc < NCLS; ++cc) mx = fmaxf(mx, fuse[row][cc]);
    float se = 0.f;
    for (int cc = 0; cc < NCLS; ++cc) se += expf(fuse[row][cc] - mx);
    float lseF = mx + logf(se);
    const float* lcr = lc + (size_t)(rbase + row) * NCLS;
    float mx2 = -3.4e38f;
    for (int cc = 0; cc < NCLS; ++cc) mx2 = fmaxf(mx2, lcr[cc]);
    float se2 = 0.f;
    for (int cc = 0; cc < NCLS; ++cc) se2 += expf(lcr[cc] - mx2);
    float lseL = mx2 + logf(se2);
    float* orow = out + (size_t)(rbase + row) * NCLS;
    for (int cc = 0; cc < NCLS; ++cc)
      orow[cc] = 0.5f * (lcr[cc] - lseL) + 0.5f * (fuse[row][cc] - lseF);
  }
}

// ---------------- launch ----------------
extern "C" void kernel_launch(void* const* d_in, const int* in_sizes, int n_in,
                              void* d_out, int out_size, void* d_ws, size_t ws_size,
                              hipStream_t stream) {
  const float* x    = (const float*)d_in[0];
  const int*   edge = (const int*)d_in[1];
  const int*   y    = (const int*)d_in[2];
  const float* mask = (const float*)d_in[3];
  const float* W1l  = (const float*)d_in[4];
  const float* b1   = (const float*)d_in[5];
  const float* W1r  = (const float*)d_in[6];
  const float* W2l  = (const float*)d_in[7];
  const float* b2   = (const float*)d_in[8];
  const float* W2r  = (const float*)d_in[9];
  const int* src = edge;
  const int* dst = edge + N_EDGES;

  char* p = (char*)d_ws;
  auto carve = [&](size_t bytes) -> char* {
    char* q = (char*)(((uintptr_t)p + 255) & ~(uintptr_t)255);
    p = q + bytes;
    return q;
  };
  int*      cnt    = (int*)carve((size_t)N_NODES * 4);
  int*      rstart = (int*)carve((size_t)(N_NODES + 1) * 4);
  int*      fpos   = (int*)carve((size_t)N_NODES * 4);
  int*      csr    = (int*)carve((size_t)N_EDGES * 4);
  _Float16* xh     = (_Float16*)carve((size_t)N_NODES * D_INF * 2);
  _Float16* a1     = (_Float16*)carve((size_t)N_NODES * D_INF * 2);
  _Float16* hh     = (_Float16*)carve((size_t)N_NODES * D_HID * 2);
  _Float16* a2     = (_Float16*)carve((size_t)N_NODES * D_HID * 2);
  _Float16* enb    = (_Float16*)carve((size_t)N_NODES * D_HID * 2);
  _Float16* w1l    = (_Float16*)carve((size_t)D_HID * D_INF * 2);
  _Float16* w1r    = (_Float16*)carve((size_t)D_HID * D_INF * 2);
  _Float16* w2l    = (_Float16*)carve((size_t)NCLS_P * D_HID * 2);
  _Float16* w2r    = (_Float16*)carve((size_t)NCLS_P * D_HID * 2);
  float*    b2p    = (float*)carve((size_t)NCLS_P * 4);
  float*    lc     = (float*)carve((size_t)N_NODES * NCLS * 4);

  float* outp = (float*)d_out;
  float* hbuf = outp + (size_t)N_NODES * NCLS;   // second tuple output: h [N, 256]

  // graph structure (CSR)
  zero_i32 <<<(N_NODES + 255) / 256, 256, 0, stream>>>(cnt, N_NODES);
  count_deg<<<(N_EDGES + 255) / 256, 256, 0, stream>>>(dst, cnt, N_EDGES);
  scan10k  <<<1, 256, 0, stream>>>(cnt, rstart, fpos, N_NODES);
  fill_csr <<<(N_EDGES + 255) / 256, 256, 0, stream>>>(src, dst, fpos, csr, N_EDGES);

  // f16 conversions
  f2h<<<((size_t)N_NODES * D_INF + 255) / 256, 256, 0, stream>>>(x, xh, N_NODES * D_INF);
  f2h<<<(D_HID * D_INF + 255) / 256, 256, 0, stream>>>(W1l, w1l, D_HID * D_INF);
  f2h<<<(D_HID * D_INF + 255) / 256, 256, 0, stream>>>(W1r, w1r, D_HID * D_INF);
  pad_w<<<(NCLS_P * D_HID + 255) / 256, 256, 0, stream>>>(W2l, w2l, NCLS, D_HID, NCLS_P * D_HID);
  pad_w<<<(NCLS_P * D_HID + 255) / 256, 256, 0, stream>>>(W2r, w2r, NCLS, D_HID, NCLS_P * D_HID);
  pad_b<<<1, 64, 0, stream>>>(b2, b2p, NCLS, NCLS_P);

  // layer 1: h = ELU(mean_agg(x)@W1l^T + b1 + x@W1r^T)
  aggregate<<<N_NODES, 256, 0, stream>>>(x, rstart, csr, a1, D_INF);
  conv_wmma<<<1250, 256, 0, stream>>>(a1, w1l, xh, w1r, b1, hbuf,
                                      D_INF, D_HID / 16, D_HID, D_HID, 10000);
  rownorm<<<N_NODES, 256, 0, stream>>>(hbuf, hh, enb);

  // layer 2: lc = ELU(mean_agg(h)@W2l^T + b2 + h@W2r^T)
  aggregate<<<N_NODES, 256, 0, stream>>>(hbuf, rstart, csr, a2, D_HID);
  conv_wmma<<<(NT_COL * 3 + 7) / 8, 256, 0, stream>>>(a2, w2l, hh, w2r, b2p, lc,
                                                      D_HID, NCLS_P / 16, NCLS, NCLS, NT_COL * 3);

  // fused similarity / top-k / label fusion / log-softmax blend
  sim_topk<<<NT_COL, 256, 0, stream>>>(enb, mask, y, lc, outp);
}
